// RoIHeads_4105988735293
// MI455X (gfx1250) — compile-verified, matrix operand order checked by reading, not run on hardware
//
#include <hip/hip_runtime.h>
#include <stdint.h>

typedef _Float16 f16;
typedef _Float16 v16h __attribute__((ext_vector_type(16)));
typedef _Float16 v8h  __attribute__((ext_vector_type(8)));
typedef float    v8f  __attribute__((ext_vector_type(8)));

#define NB   4
#define NR   512
#define NROI 2048          // NB*NR
#define CF   256
#define DIN  12544         // 256*49
#define HID  1024
#define NC   91
#define NCP  96
#define NR4  364
#define NR4P 368
#define TOPK 2048
#define NSORT 65536
#define NCAND 46080        // 512*90
#define DETS 100
#define BBOX_CLIP 4.135166556742356f

// ---------------- features NCHW -> NHWC ----------------
__global__ void k_transpose_feat(const float* __restrict__ f, float* __restrict__ fT) {
    int p = blockIdx.x;            // b*2500 + y*50 + x
    int c = threadIdx.x;           // 0..255
    int b = p / 2500;
    int yx = p % 2500;
    fT[(long)p * CF + c] = f[((long)(b * CF + c)) * 2500 + yx];
}

// -------- weight convert: f32 [K][N] -> f16 [Npad][K] (optional RoI K-permute) --------
__global__ void k_convert_weight(const float* __restrict__ W, f16* __restrict__ Wt,
                                 int K, int N, int Npad, int roi_perm) {
    long tid = (long)blockIdx.x * blockDim.x + threadIdx.x;
    long total = (long)Npad * K;
    if (tid >= total) return;
    int kd = (int)(tid % K);
    int nd = (int)(tid / K);
    float v = 0.f;
    if (nd < N) {
        int ks = kd;
        if (roi_perm) { int c = kd & 255; int pos = kd >> 8; ks = c * 49 + pos; }
        v = W[(long)ks * N + nd];
    }
    Wt[tid] = (f16)v;
}

// ---------------- RoIAlign -> A f16 [2048][12544], K index = pos*256+c ----------------
__global__ void k_roi_align(const float* __restrict__ fT, const float* __restrict__ props,
                            f16* __restrict__ A) {
    int blk = blockIdx.x;          // roi*49 + pos
    int pos = blk % 49;
    int roi = blk / 49;            // b*512 + r
    int b   = roi >> 9;
    int py = pos / 7, px = pos % 7;
    const float* pr = props + (long)roi * 4;
    float x1 = pr[0] * 0.0625f, y1 = pr[1] * 0.0625f;
    float x2 = pr[2] * 0.0625f, y2 = pr[3] * 0.0625f;
    float bh = (y2 - y1) * (1.f / 7.f), bw = (x2 - x1) * (1.f / 7.f);
    float y = y1 + ((float)py + 0.5f) * bh;
    float x = x1 + ((float)px + 0.5f) * bw;
    y = fminf(fmaxf(y, 0.f), 49.f);
    x = fminf(fmaxf(x, 0.f), 49.f);
    float y0f = floorf(y), x0f = floorf(x);
    int y0 = (int)y0f, x0 = (int)x0f;
    int y1i = (y0 + 1 > 49) ? 49 : y0 + 1;
    int x1i = (x0 + 1 > 49) ? 49 : x0 + 1;
    float ly = y - y0f, lx = x - x0f, hy = 1.f - ly, hx = 1.f - lx;
    float w00 = hy * hx, w01 = hy * lx, w10 = ly * hx, w11 = ly * lx;
    int c = threadIdx.x;
    long base = (long)b * 2500 * CF;
    float f00 = fT[base + (long)(y0  * 50 + x0 ) * CF + c];
    float f01 = fT[base + (long)(y0  * 50 + x1i) * CF + c];
    float f10 = fT[base + (long)(y1i * 50 + x0 ) * CF + c];
    float f11 = fT[base + (long)(y1i * 50 + x1i) * CF + c];
    float out = w00 * f00 + w01 * f01 + w10 * f10 + w11 * f11;
    A[(long)roi * DIN + pos * CF + c] = (f16)out;
}

// ---------------- WMMA GEMM: C[M,N] = A[M,K](f16) * Bt[Npad,K](f16) + bias ----------------
// 256 threads (8 wave32), C tile 128x128, wave tile 64x32 (4x2 wmma accs),
// double-buffered LDS => single barrier per K-step; GUARDN only for head GEMMs.
template<bool RELU, bool F16OUT, bool GUARDN>
__global__ __launch_bounds__(256) void k_gemm(const f16* __restrict__ A, const f16* __restrict__ Bt,
                                              const float* __restrict__ bias,
                                              float* __restrict__ Cf, f16* __restrict__ Ch,
                                              int N, int K, int Npad, int ldc) {
    __shared__ __align__(16) f16 As[2][128][40];   // 32 K + 8 pad (conflict-free frag loads)
    __shared__ __align__(16) f16 Bs[2][128][40];
    int tid  = threadIdx.x;
    int lane = tid & 31;
    int wave = tid >> 5;
    int waveM = (wave >> 2) * 64;
    int waveN = (wave & 3) * 32;
    int blockM = blockIdx.y * 128;
    int blockN = blockIdx.x * 128;
    int halfsel = lane >> 4;       // K-subgroup select per ISA 16-bit A-fragment layout
    int lrow = lane & 15;
    int koff = halfsel * 8;

    // staging thread -> (row, sub) mapping for the two 16B chunks it owns
    int row0 = tid >> 2,        sub0 = tid & 3;
    int row1 = (tid + 256) >> 2, sub1 = tid & 3;   // (tid+256)&3 == tid&3

    v8f acc[4][2];
    for (int i = 0; i < 4; i++)
        for (int j = 0; j < 2; j++)
            for (int r = 0; r < 8; r++) acc[i][j][r] = 0.f;

    v8h a0, a1, b0, b1;
    auto loadTile = [&](int kt) {
        a0 = *(const v8h*)(A + (long)(blockM + row0) * K + kt + sub0 * 8);
        a1 = *(const v8h*)(A + (long)(blockM + row1) * K + kt + sub1 * 8);
        if (GUARDN) {
            v8h z; for (int e = 0; e < 8; e++) z[e] = (f16)0.f;
            b0 = (blockN + row0 < Npad) ? *(const v8h*)(Bt + (long)(blockN + row0) * K + kt + sub0 * 8) : z;
            b1 = (blockN + row1 < Npad) ? *(const v8h*)(Bt + (long)(blockN + row1) * K + kt + sub1 * 8) : z;
        } else {
            b0 = *(const v8h*)(Bt + (long)(blockN + row0) * K + kt + sub0 * 8);
            b1 = *(const v8h*)(Bt + (long)(blockN + row1) * K + kt + sub1 * 8);
        }
    };

    loadTile(0);
    for (int kt = 0; kt < K; kt += 32) {
        int buf = (kt >> 5) & 1;
        *(v8h*)&As[buf][row0][sub0 * 8] = a0;
        *(v8h*)&As[buf][row1][sub1 * 8] = a1;
        *(v8h*)&Bs[buf][row0][sub0 * 8] = b0;
        *(v8h*)&Bs[buf][row1][sub1 * 8] = b1;
        __syncthreads();

        if (kt + 32 < K) loadTile(kt + 32);
        if (kt + 64 < K) {      // warm GL2 two tiles ahead -> global_prefetch_b8
            __builtin_prefetch(A + (long)(blockM + row0) * K + kt + 64, 0, 3);
            if (!GUARDN || blockN + row0 < Npad)
                __builtin_prefetch(Bt + (long)(blockN + row0) * K + kt + 64, 0, 3);
        }

        v16h af[4], bf[2];
        for (int mi = 0; mi < 4; mi++) {
            int r = waveM + mi * 16 + lrow;
            v8h lo = *(const v8h*)&As[buf][r][koff];
            v8h hi = *(const v8h*)&As[buf][r][koff + 16];
            af[mi] = __builtin_shufflevector(lo, hi, 0,1,2,3,4,5,6,7,8,9,10,11,12,13,14,15);
        }
        for (int ni = 0; ni < 2; ni++) {
            int r = waveN + ni * 16 + lrow;
            v8h lo = *(const v8h*)&Bs[buf][r][koff];
            v8h hi = *(const v8h*)&Bs[buf][r][koff + 16];
            bf[ni] = __builtin_shufflevector(lo, hi, 0,1,2,3,4,5,6,7,8,9,10,11,12,13,14,15);
        }
        for (int mi = 0; mi < 4; mi++)
            for (int ni = 0; ni < 2; ni++)
                acc[mi][ni] = __builtin_amdgcn_wmma_f32_16x16x32_f16(
                    false, af[mi], false, bf[ni], (short)0, acc[mi][ni], false, false);
    }

    for (int mi = 0; mi < 4; mi++) {
        for (int ni = 0; ni < 2; ni++) {
            int col = blockN + waveN + ni * 16 + lrow;
            if (col >= N) continue;               // N = real N
            float bv = bias[col];
            int mbase = blockM + waveM + mi * 16 + halfsel * 8;
            for (int r = 0; r < 8; r++) {
                float v = acc[mi][ni][r] + bv;
                if (RELU) v = fmaxf(v, 0.f);
                long off = (long)(mbase + r) * ldc + col;
                if (F16OUT) Ch[off] = (f16)v; else Cf[off] = v;
            }
        }
    }
}

// ---------------- softmax over 91 classes ----------------
__global__ void k_softmax(const float* __restrict__ logits, float* __restrict__ scores) {
    __shared__ float red[128];
    int row = blockIdx.x, t = threadIdx.x;
    float v = (t < NC) ? logits[(long)row * NC + t] : -3.4e38f;
    red[t] = v; __syncthreads();
    for (int s = 64; s > 0; s >>= 1) { if (t < s) red[t] = fmaxf(red[t], red[t + s]); __syncthreads(); }
    float mx = red[0]; __syncthreads();
    float e = (t < NC) ? expf(v - mx) : 0.f;
    red[t] = e; __syncthreads();
    for (int s = 64; s > 0; s >>= 1) { if (t < s) red[t] += red[t + s]; __syncthreads(); }
    float sum = red[0];
    if (t < NC) scores[(long)row * NC + t] = e / sum;
}

// ---------------- box decode + clip ----------------
__global__ void k_decode(const float* __restrict__ breg, const float* __restrict__ props,
                         float* __restrict__ bdec) {
    int row = blockIdx.x;
    int cls = threadIdx.x;
    if (cls >= NC) return;
    const float* pr = props + (long)row * 4;
    float w = pr[2] - pr[0], h = pr[3] - pr[1];
    float cx = pr[0] + 0.5f * w, cy = pr[1] + 0.5f * h;
    const float* d = breg + (long)row * NR4 + cls * 4;
    float dx = d[0] * 0.1f, dy = d[1] * 0.1f;
    float dw = fminf(d[2] * 0.2f, BBOX_CLIP);
    float dh = fminf(d[3] * 0.2f, BBOX_CLIP);
    float pcx = dx * w + cx, pcy = dy * h + cy;
    float pw = expf(dw) * w, ph = expf(dh) * h;
    float x1 = fminf(fmaxf(pcx - 0.5f * pw, 0.f), 800.f);
    float yy1 = fminf(fmaxf(pcy - 0.5f * ph, 0.f), 800.f);
    float x2 = fminf(fmaxf(pcx + 0.5f * pw, 0.f), 800.f);
    float yy2 = fminf(fmaxf(pcy + 0.5f * ph, 0.f), 800.f);
    float* o = bdec + ((long)row * NC + cls) * 4;
    o[0] = x1; o[1] = yy1; o[2] = x2; o[3] = yy2;
}

// ---------------- candidate keys (monotone float->uint, desc score, idx tiebreak) ----------------
__device__ inline unsigned f2ord(float f) {
    unsigned u = __float_as_uint(f);
    return (u & 0x80000000u) ? ~u : (u | 0x80000000u);
}

__global__ void k_build_keys(const float* __restrict__ scores, const float* __restrict__ bdec,
                             unsigned long long* __restrict__ keys) {
    int tid = blockIdx.x * 256 + threadIdx.x;   // B*65536 threads
    int b = tid >> 16;
    int j = tid & (NSORT - 1);
    unsigned long long key = ~0ull;
    if (j < NCAND) {
        int roi = j / 90, cls = j % 90 + 1;
        int row = b * NR + roi;
        float s = scores[(long)row * NC + cls];
        const float* bx = bdec + ((long)row * NC + cls) * 4;
        float w = bx[2] - bx[0], h = bx[3] - bx[1];
        int valid = (s > 0.05f) && (w >= 0.01f) && (h >= 0.01f);
        float cand = valid ? s : -1.f;
        key = ((unsigned long long)(~f2ord(cand)) << 32) | (unsigned)j;
    }
    keys[tid] = key;
}

__global__ void k_bitonic(unsigned long long* __restrict__ keys, int j, int k) {
    int tid = blockIdx.x * 256 + threadIdx.x;   // B*65536 threads
    int b = tid >> 16;
    int i = tid & (NSORT - 1);
    int ixj = i ^ j;
    if (ixj <= i) return;
    unsigned long long* base = keys + ((long)b << 16);
    unsigned long long a = base[i], c = base[ixj];
    bool up = ((i & k) == 0);
    if ((a > c) == up) { base[i] = c; base[ixj] = a; }
}

// ---------------- gather sorted top-2048 ----------------
__global__ void k_gather_topk(const unsigned long long* __restrict__ keys,
                              const float* __restrict__ scores, const float* __restrict__ bdec,
                              float* __restrict__ tkbox, float* __restrict__ tkscore,
                              int* __restrict__ tklabel, int* __restrict__ tkvalid) {
    int b = blockIdx.x;
    int k = blockIdx.y * 256 + threadIdx.x;     // 0..2047
    unsigned idx = (unsigned)(keys[((long)b << 16) + k] & 0xFFFFFFFFu);
    float bx0 = 0.f, bx1 = 0.f, bx2 = 0.f, bx3 = 0.f, s = 0.f;
    int lab = 0, valid = 0;
    if (idx < (unsigned)NCAND) {
        int roi = idx / 90, cls = idx % 90 + 1;
        int row = b * NR + roi;
        s = scores[(long)row * NC + cls];
        const float* p = bdec + ((long)row * NC + cls) * 4;
        bx0 = p[0]; bx1 = p[1]; bx2 = p[2]; bx3 = p[3];
        float w = bx2 - bx0, h = bx3 - bx1;
        valid = (s > 0.05f) && (w >= 0.01f) && (h >= 0.01f);
        lab = cls;
    }
    long o = (long)b * TOPK + k;
    tkbox[o * 4 + 0] = bx0; tkbox[o * 4 + 1] = bx1;
    tkbox[o * 4 + 2] = bx2; tkbox[o * 4 + 3] = bx3;
    tkscore[o] = s; tklabel[o] = lab; tkvalid[o] = valid;
}

// ---------------- NMS suppression bitmask ----------------
__global__ void k_nms_mask(const float* __restrict__ tkbox, const int* __restrict__ tklabel,
                           unsigned long long* __restrict__ mask) {
    int b = blockIdx.z, ib = blockIdx.y, jb = blockIdx.x;
    int t = threadIdx.x;                        // 64
    __shared__ float jbox[64][4];
    __shared__ int jlab[64];
    long base = (long)b * TOPK;
    int jg = jb * 64 + t;
    jbox[t][0] = tkbox[(base + jg) * 4 + 0];
    jbox[t][1] = tkbox[(base + jg) * 4 + 1];
    jbox[t][2] = tkbox[(base + jg) * 4 + 2];
    jbox[t][3] = tkbox[(base + jg) * 4 + 3];
    jlab[t] = tklabel[base + jg];
    __syncthreads();
    int i = ib * 64 + t;
    float ax1 = tkbox[(base + i) * 4 + 0], ay1 = tkbox[(base + i) * 4 + 1];
    float ax2 = tkbox[(base + i) * 4 + 2], ay2 = tkbox[(base + i) * 4 + 3];
    int al = tklabel[base + i];
    float aarea = (ax2 - ax1) * (ay2 - ay1);
    unsigned long long bits = 0;
    for (int jj = 0; jj < 64; jj++) {
        int j = jb * 64 + jj;
        if (j <= i || jlab[jj] != al) continue;
        float ix1 = fmaxf(ax1, jbox[jj][0]), iy1 = fmaxf(ay1, jbox[jj][1]);
        float ix2 = fminf(ax2, jbox[jj][2]), iy2 = fminf(ay2, jbox[jj][3]);
        float iw = fmaxf(ix2 - ix1, 0.f), ih = fmaxf(iy2 - iy1, 0.f);
        float inter = iw * ih;
        float barea = (jbox[jj][2] - jbox[jj][0]) * (jbox[jj][3] - jbox[jj][1]);
        float iou = inter / (aarea + barea - inter);
        if (iou > 0.5f) bits |= (1ull << jj);
    }
    mask[(base + i) * 32 + jb] = bits;
}

// ---------------- serial greedy scan (one wave32 per image) + top-100 compact ----------------
__global__ void k_nms_scan(const unsigned long long* __restrict__ mask,
                           const int* __restrict__ tkvalid,
                           const float* __restrict__ tkbox, const float* __restrict__ tkscore,
                           const int* __restrict__ tklabel,
                           float* __restrict__ out) {
    int b = blockIdx.x;
    int lane = threadIdx.x;                     // 32 threads, lane owns 64-col word
    long base = (long)b * TOPK;
    unsigned long long remv = 0, keepw = 0;
    for (int i = 0; i < TOPK; i++) {
        int owner = i >> 6, bit = i & 63;
        int alive = 0;
        if (lane == owner) alive = tkvalid[base + i] && !((remv >> bit) & 1);
        alive = __shfl(alive, owner, 32);
        if (alive) {
            remv |= mask[(base + i) * 32 + lane];
            if (lane == owner) keepw |= (1ull << bit);
        }
    }
    __shared__ unsigned long long kw[32];
    kw[lane] = keepw;
    __syncthreads();
    for (int q = lane; q < DETS * 6; q += 32) {
        if (q < DETS * 4)       out[b * DETS * 4 + q] = 0.f;
        else if (q < DETS * 5)  out[NB * DETS * 4 + b * DETS + (q - DETS * 4)] = 0.f;
        else                    out[NB * DETS * 5 + b * DETS + (q - DETS * 5)] = 0.f;
    }
    __syncthreads();
    if (lane == 0) {
        int cnt = 0;
        for (int i = 0; i < TOPK && cnt < DETS; i++) {
            if ((kw[i >> 6] >> (i & 63)) & 1) {
                out[b * DETS * 4 + cnt * 4 + 0] = tkbox[(base + i) * 4 + 0];
                out[b * DETS * 4 + cnt * 4 + 1] = tkbox[(base + i) * 4 + 1];
                out[b * DETS * 4 + cnt * 4 + 2] = tkbox[(base + i) * 4 + 2];
                out[b * DETS * 4 + cnt * 4 + 3] = tkbox[(base + i) * 4 + 3];
                out[NB * DETS * 4 + b * DETS + cnt] = tkscore[base + i];
                out[NB * DETS * 5 + b * DETS + cnt] = (float)tklabel[base + i];
                cnt++;
            }
        }
    }
}

extern "C" void kernel_launch(void* const* d_in, const int* in_sizes, int n_in,
                              void* d_out, int out_size, void* d_ws, size_t ws_size,
                              hipStream_t stream) {
    const float* features  = (const float*)d_in[0];
    const float* proposals = (const float*)d_in[1];
    const float* W1 = (const float*)d_in[2];
    const float* b1 = (const float*)d_in[3];
    const float* W2 = (const float*)d_in[4];
    const float* b2 = (const float*)d_in[5];
    const float* Wc = (const float*)d_in[6];
    const float* bc = (const float*)d_in[7];
    const float* Wr = (const float*)d_in[8];
    const float* br = (const float*)d_in[9];
    float* out = (float*)d_out;
    (void)in_sizes; (void)n_in; (void)out_size; (void)ws_size;

    char* ws = (char*)d_ws;
    size_t off = 0;
    auto carve = [&](size_t bytes) -> void* {
        size_t o = (off + 255) & ~(size_t)255;
        off = o + bytes;
        return (void*)(ws + o);
    };
    float* featT  = (float*)carve((size_t)NB * 2500 * CF * 4);
    f16*   Aroi   = (f16*)  carve((size_t)NROI * DIN * 2);
    f16*   W1t    = (f16*)  carve((size_t)HID * DIN * 2);
    f16*   W2t    = (f16*)  carve((size_t)HID * HID * 2);
    f16*   Wct    = (f16*)  carve((size_t)NCP * HID * 2);
    f16*   Wrt    = (f16*)  carve((size_t)NR4P * HID * 2);
    f16*   H1     = (f16*)  carve((size_t)NROI * HID * 2);
    f16*   H2     = (f16*)  carve((size_t)NROI * HID * 2);
    float* logits = (float*)carve((size_t)NROI * NC * 4);
    float* scores = (float*)carve((size_t)NROI * NC * 4);
    float* breg   = (float*)carve((size_t)NROI * NR4 * 4);
    float* bdec   = (float*)carve((size_t)NROI * NC * 4 * 4);
    unsigned long long* keys = (unsigned long long*)carve((size_t)NB * NSORT * 8);
    float* tkbox   = (float*)carve((size_t)NB * TOPK * 4 * 4);
    float* tkscore = (float*)carve((size_t)NB * TOPK * 4);
    int*   tklabel = (int*)  carve((size_t)NB * TOPK * 4);
    int*   tkvalid = (int*)  carve((size_t)NB * TOPK * 4);
    unsigned long long* nmask = (unsigned long long*)carve((size_t)NB * TOPK * 32 * 8);

    // 1) feature transpose + weight conversion
    k_transpose_feat<<<NB * 2500, 256, 0, stream>>>(features, featT);
    k_convert_weight<<<((size_t)HID * DIN + 255) / 256, 256, 0, stream>>>(W1, W1t, DIN, HID, HID, 1);
    k_convert_weight<<<((size_t)HID * HID + 255) / 256, 256, 0, stream>>>(W2, W2t, HID, HID, HID, 0);
    k_convert_weight<<<((size_t)NCP * HID + 255) / 256, 256, 0, stream>>>(Wc, Wct, HID, NC, NCP, 0);
    k_convert_weight<<<((size_t)NR4P * HID + 255) / 256, 256, 0, stream>>>(Wr, Wrt, HID, NR4, NR4P, 0);

    // 2) RoIAlign -> f16 activation matrix
    k_roi_align<<<NROI * 49, 256, 0, stream>>>(featT, proposals, Aroi);

    // 3) WMMA GEMM chain (branchless inner loop for the two big GEMMs)
    k_gemm<true,  true,  false><<<dim3(HID / 128, NROI / 128), 256, 0, stream>>>(Aroi, W1t, b1, nullptr, H1, HID, DIN, HID, HID);
    k_gemm<true,  true,  false><<<dim3(HID / 128, NROI / 128), 256, 0, stream>>>(H1,   W2t, b2, nullptr, H2, HID, HID, HID, HID);
    k_gemm<false, false, true ><<<dim3(1,         NROI / 128), 256, 0, stream>>>(H2,   Wct, bc, logits, nullptr, NC,  HID, NCP,  NC);
    k_gemm<false, false, true ><<<dim3(3,         NROI / 128), 256, 0, stream>>>(H2,   Wrt, br, breg,   nullptr, NR4, HID, NR4P, NR4);

    // 4) softmax + box decode
    k_softmax<<<NROI, 128, 0, stream>>>(logits, scores);
    k_decode<<<NROI, 96, 0, stream>>>(breg, proposals, bdec);

    // 5) candidate keys + bitonic sort (desc score, idx tiebreak)
    k_build_keys<<<NB * NSORT / 256, 256, 0, stream>>>(scores, bdec, keys);
    for (int k = 2; k <= NSORT; k <<= 1)
        for (int j = k >> 1; j > 0; j >>= 1)
            k_bitonic<<<NB * NSORT / 256, 256, 0, stream>>>(keys, j, k);

    // 6) gather top-2048, bitmask NMS, compact top-100
    k_gather_topk<<<dim3(NB, TOPK / 256), 256, 0, stream>>>(keys, scores, bdec, tkbox, tkscore, tklabel, tkvalid);
    k_nms_mask<<<dim3(TOPK / 64, TOPK / 64, NB), 64, 0, stream>>>(tkbox, tklabel, nmask);
    k_nms_scan<<<NB, 32, 0, stream>>>(nmask, tkvalid, tkbox, tkscore, tklabel, out);
}
